// LanguageModelSRNN_35029753266491
// MI455X (gfx1250) — matrix operand order repaired and checked by reading, not compiled, not run on hardware
//
#include <hip/hip_runtime.h>
#include <hip/hip_bf16.h>
#include <math.h>

// CDNA5 wave32 WMMA vector types
typedef __attribute__((ext_vector_type(16))) __bf16 v16bf;
typedef __attribute__((ext_vector_type(8)))  __bf16 v8bf;
typedef __attribute__((ext_vector_type(8)))  float  v8f;

#define B_  16
#define S_  256
#define H_  512
#define V_  32000

__device__ __forceinline__ unsigned short f2bf(float f) {
    unsigned int u = __float_as_uint(f);
    unsigned int r = u + 0x7FFFu + ((u >> 16) & 1u);   // round-to-nearest-even
    return (unsigned short)(r >> 16);
}

// low 32 bits of a generic pointer to a __shared__ object == LDS byte offset
__device__ __forceinline__ unsigned lds_off(const void* p) {
    return (unsigned)(unsigned long long)p;
}

// Async copy 16B global -> LDS (gfx1250 ASYNCcnt path).
__device__ __forceinline__ void async_ld_b128(unsigned ldsByteOff, const void* g) {
    asm volatile("global_load_async_to_lds_b128 %0, %1, off"
                 :: "v"(ldsByteOff), "v"(g) : "memory");
}
__device__ __forceinline__ void async_ld_b128_off16(unsigned ldsByteOff, const void* g) {
    asm volatile("global_load_async_to_lds_b128 %0, %1, off offset:16"
                 :: "v"(ldsByteOff), "v"(g) : "memory");
}
__device__ __forceinline__ void wait_async_le3() {
    asm volatile("s_wait_asynccnt 0x3" ::: "memory");
}
__device__ __forceinline__ void wait_async_0() {
    asm volatile("s_wait_asynccnt 0x0" ::: "memory");
}

// ---------------------------------------------------------------- fp32 -> bf16
__global__ void f32_to_bf16_kernel(const float* __restrict__ in,
                                   unsigned short* __restrict__ out, int n) {
    int i = blockIdx.x * blockDim.x + threadIdx.x;
    if (i < n) out[i] = f2bf(in[i]);
}

// -------------------------------------------------- emb[t][b][j] = W[j][X[b,t]]
__global__ void gather_emb_kernel(const float* __restrict__ W,
                                  const int* __restrict__ X,
                                  float* __restrict__ emb) {
    int i = blockIdx.x * blockDim.x + threadIdx.x;   // < S*B*H = 2097152
    int j = i & (H_ - 1);
    int b = (i >> 9) & (B_ - 1);
    int t = i >> 13;
    int x = X[b * S_ + t];
    emb[i] = W[(size_t)j * V_ + x];
}

// ------------------------------------------------------------------ recurrence
// One workgroup (32 waves) owns the whole sequential scan. Wave w computes the
// 16 columns [16w,16w+16) of h_new via 16x v_wmma_f32_16x16x32_bf16 (K=512).
#define HSTR 520
__global__ __launch_bounds__(1024) void rnn_recurrence_kernel(
    const unsigned short* __restrict__ Ub,   // (H,H) bf16 row-major
    const float* __restrict__ emb,           // (S,B,H) fp32
    unsigned short* __restrict__ h_all)      // (B*S, H) bf16, row = b*S + t
{
    __shared__ unsigned short hcur[B_ * HSTR];   // 16.25 KB, stride-520 anti-conflict

    const int tid   = threadIdx.x;
    const int lane  = tid & 31;
    const int wave  = tid >> 5;        // 0..31
    const int col16 = lane & 15;
    const int half  = lane >> 4;       // 0|1
    const int jbase = wave * 16;

    for (int i = tid; i < B_ * HSTR; i += 1024) hcur[i] = 0;  // h0 = 0
    __syncthreads();

    const int arow = col16;
    const int akb  = half * 16;
    const unsigned short* brow = &Ub[(size_t)(jbase + col16) * H_ + half * 16];

    for (int t = 0; t < S_; ++t) {
        // kill cross-iteration caching of U fragments (was spilling to scratch);
        // U streams from L2/WGP$ every step instead.
        asm volatile("" ::: "memory");
        v8f acc = {};
        #pragma unroll
        for (int ks = 0; ks < 16; ++ks) {
            const int k0 = ks * 32;
            union { v16bf v; v8bf h[2]; } A;
            A.h[0] = *(const v8bf*)(&hcur[arow * HSTR + k0 + akb]);
            A.h[1] = *(const v8bf*)(&hcur[arow * HSTR + k0 + 8 + akb]);
            v16bf Bv = *(const v16bf*)(brow + k0);
            acc = __builtin_amdgcn_wmma_f32_16x16x32_bf16(
                      false, A.v, false, Bv, (short)0, acc, false, false);
        }
        __syncthreads();   // everyone done reading old hcur
        const float* et = &emb[(size_t)t * B_ * H_];
        #pragma unroll
        for (int r = 0; r < 8; ++r) {
            const int m = r + half * 8;
            const int n = jbase + col16;
            float v = tanhf(acc[r] + et[m * H_ + n]);
            unsigned short bv = f2bf(v);
            hcur[m * HSTR + n] = bv;
            h_all[((size_t)m * S_ + t) * H_ + n] = bv;
        }
        __syncthreads();
    }
}

// ------------------------------------------------------------- logits = H x V^T
// Block tile 128(M) x 256(N), 512 threads = 16 waves (4x4). Wave computes a
// 32x64 macro tile = 2x4 WMMA accumulators. K staged in 32-deep slices through
// double-buffered LDS via global_load_async_to_lds_b128 (ASYNCcnt pipelined).
#define KSLC   32
#define APAD   40                      // 32 halfs + 8 pad -> conflict-free b128
#define A_ELE  (128 * APAD)            // 5120 halfs / buffer
#define B_ELE  (256 * APAD)            // 10240 halfs / buffer
__global__ __launch_bounds__(512) void logits_gemm_kernel(
    const unsigned short* __restrict__ Hb,   // (4096,512) bf16
    const unsigned short* __restrict__ Vb,   // (32000,512) bf16
    float* __restrict__ out)                 // (4096,32000) fp32 logits
{
    __shared__ unsigned short Abuf[2][A_ELE];   // 2 x 10 KB
    __shared__ unsigned short Bbuf[2][B_ELE];   // 2 x 20 KB

    const int tid   = threadIdx.x;
    const int lane  = tid & 31;
    const int wave  = tid >> 5;            // 0..15
    const int col16 = lane & 15;
    const int half  = lane >> 4;
    const int wr    = wave >> 2;           // 0..3  (M direction)
    const int wc    = wave & 3;            // 0..3  (N direction)
    const int mbase = blockIdx.x * 128;
    const int nbase = blockIdx.y * 256;

    // ---- staging maps (per thread, per stage = 1 A b128 + 2 B b128 = 3 async)
    const int ar = tid >> 2, ac = tid & 3;          // A: 128 rows x 4 x 16B
    const int br = tid >> 1, bc = tid & 1;          // B: 256 rows x 2 x 32B
    const unsigned short* aG = &Hb[(size_t)(mbase + ar) * H_ + ac * 8];
    const unsigned short* bG = &Vb[(size_t)(nbase + br) * H_ + bc * 16];
    unsigned aL0 = lds_off(&Abuf[0][ar * APAD + ac * 8]);
    unsigned aL1 = lds_off(&Abuf[1][ar * APAD + ac * 8]);
    unsigned bL0 = lds_off(&Bbuf[0][br * APAD + bc * 16]);
    unsigned bL1 = lds_off(&Bbuf[1][br * APAD + bc * 16]);

    // ---- compute maps
    const int aRow = wr * 32 + col16;      // + mt*16
    const int bRow = wc * 64 + col16;      // + nt*16
    const int akb  = half * 16;
    const int bkb  = half * 16;

    v8f acc[2][4];
    #pragma unroll
    for (int mt = 0; mt < 2; ++mt)
        #pragma unroll
        for (int nt = 0; nt < 4; ++nt) acc[mt][nt] = (v8f){};

    // prologue: stage k-slice 0 into buffer 0
    async_ld_b128(aL0, aG);
    async_ld_b128(bL0, bG);
    async_ld_b128_off16(bL0, bG);

    for (int ks = 0; ks < 16; ++ks) {
        const int cur = ks & 1;
        if (ks + 1 < 16) {               // stage next slice into other buffer
            const int k1 = (ks + 1) * KSLC;
            async_ld_b128(cur ? aL0 : aL1, aG + k1);
            async_ld_b128(cur ? bL0 : bL1, bG + k1);
            async_ld_b128_off16(cur ? bL0 : bL1, bG + k1);
            wait_async_le3();            // current slice landed, next in flight
        } else {
            wait_async_0();
        }
        __syncthreads();

        const unsigned short* Ab = &Abuf[cur][0];
        const unsigned short* Bb = &Bbuf[cur][0];
        union { v16bf v; v8bf h[2]; } A[2];
        #pragma unroll
        for (int mt = 0; mt < 2; ++mt) {
            const unsigned short* ap = Ab + (aRow + mt * 16) * APAD;
            A[mt].h[0] = *(const v8bf*)(ap + akb);
            A[mt].h[1] = *(const v8bf*)(ap + 8 + akb);
        }
        #pragma unroll
        for (int nt = 0; nt < 4; ++nt) {
            v16bf Bv = *(const v16bf*)(Bb + (bRow + nt * 16) * APAD + bkb);
            #pragma unroll
            for (int mt = 0; mt < 2; ++mt)
                acc[mt][nt] = __builtin_amdgcn_wmma_f32_16x16x32_bf16(
                                  false, A[mt].v, false, Bv, (short)0,
                                  acc[mt][nt], false, false);
        }
        __syncthreads();                 // compute done before buffer reuse
    }

    #pragma unroll
    for (int mt = 0; mt < 2; ++mt) {
        const int orow = mbase + wr * 32 + mt * 16 + half * 8;
        #pragma unroll
        for (int nt = 0; nt < 4; ++nt) {
            float* o = &out[(size_t)orow * V_ + nbase + wc * 64 + nt * 16 + col16];
            #pragma unroll
            for (int r = 0; r < 8; ++r) o[(size_t)r * V_] = acc[mt][nt][r];
        }
    }
}

// ------------------------------------------------------- in-place row softmax
__global__ __launch_bounds__(1024) void softmax_rows_kernel(float* __restrict__ out) {
    __shared__ float red[1024];
    float* p = out + (size_t)blockIdx.x * V_;
    const int tid = threadIdx.x;

    float vals[32];
    float mx = -3.4e38f;
    #pragma unroll
    for (int i = 0; i < 32; ++i) {
        int idx = tid + i * 1024;
        float v = (idx < V_) ? p[idx] : -3.4e38f;
        vals[i] = v;
        mx = fmaxf(mx, v);
    }
    red[tid] = mx; __syncthreads();
    for (int s = 512; s > 0; s >>= 1) {
        if (tid < s) red[tid] = fmaxf(red[tid], red[tid + s]);
        __syncthreads();
    }
    mx = red[0]; __syncthreads();

    float sum = 0.f;
    #pragma unroll
    for (int i = 0; i < 32; ++i) {
        int idx = tid + i * 1024;
        float e = (idx < V_) ? __expf(vals[i] - mx) : 0.f;
        vals[i] = e;
        sum += e;
    }
    red[tid] = sum; __syncthreads();
    for (int s = 512; s > 0; s >>= 1) {
        if (tid < s) red[tid] += red[tid + s];
        __syncthreads();
    }
    const float inv = 1.0f / red[0];
    #pragma unroll
    for (int i = 0; i < 32; ++i) {
        int idx = tid + i * 1024;
        if (idx < V_) p[idx] = vals[i] * inv;
    }
}

// ------------------------------------------------------------------- launcher
extern "C" void kernel_launch(void* const* d_in, const int* in_sizes, int n_in,
                              void* d_out, int out_size, void* d_ws, size_t ws_size,
                              hipStream_t stream) {
    const int*   X = (const int*)  d_in[0];   // (B,S)
    const float* U = (const float*)d_in[1];   // (H,H)
    const float* W = (const float*)d_in[2];   // (H,VOCAB)
    const float* V = (const float*)d_in[3];   // (VOCAB,H)
    float* out = (float*)d_out;               // (B,S,VOCAB)

    // workspace layout
    char* ws = (char*)d_ws;
    unsigned short* Ub    = (unsigned short*)(ws);                       // 512 KB
    unsigned short* Vb    = (unsigned short*)(ws + 524288);              // 32.77 MB
    float*          emb   = (float*)         (ws + 33292288);           // 8.39 MB
    unsigned short* h_all = (unsigned short*)(ws + 41680896);           // 4.19 MB

    f32_to_bf16_kernel<<<(H_ * H_ + 255) / 256, 256, 0, stream>>>(U, Ub, H_ * H_);
    f32_to_bf16_kernel<<<(V_ * H_ + 255) / 256, 256, 0, stream>>>(V, Vb, V_ * H_);
    gather_emb_kernel<<<(S_ * B_ * H_) / 256, 256, 0, stream>>>(W, X, emb);
    rnn_recurrence_kernel<<<1, 1024, 0, stream>>>(Ub, emb, h_all);
    logits_gemm_kernel<<<dim3(128 / 128 * 32, V_ / 256), 512, 0, stream>>>(h_all, Vb, out);
    softmax_rows_kernel<<<B_ * S_, 1024, 0, stream>>>(out);
}
